// LocalAttention_3298534884179
// MI455X (gfx1250) — matrix operand order repaired and checked by reading, not compiled, not run on hardware
//
#include <hip/hip_runtime.h>
#include <hip/hip_bf16.h>

// ---------- WMMA fragment types (gfx1250, wave32) ----------
typedef __attribute__((ext_vector_type(16))) __bf16 v16bf;
typedef __attribute__((ext_vector_type(8)))  __bf16 v8bf;
typedef __attribute__((ext_vector_type(8)))  float  v8f;

union V16 { v16bf v; v8bf h[2]; };

#define B_DIM   2
#define S_DIM   4096
#define D_MODEL 512
#define NHEAD   8
#define HD      64
#define D3      1536

__device__ __forceinline__ v8f wmma_bf16(v16bf a, v16bf b, v8f c) {
    return __builtin_amdgcn_wmma_f32_16x16x32_bf16(false, a, false, b, (short)0, c,
                                                   false, false);
}

__device__ __forceinline__ float rmax16(float v) {
#pragma unroll
    for (int m = 1; m < 16; m <<= 1) v = fmaxf(v, __shfl_xor(v, m, 32));
    return v;
}
__device__ __forceinline__ float rsum16(float v) {
#pragma unroll
    for (int m = 1; m < 16; m <<= 1) v += __shfl_xor(v, m, 32);
    return v;
}

// CDNA5 async global->LDS copy, 16B per lane. ldsOff = byte address in group segment.
__device__ __forceinline__ void async_ld16(unsigned ldsOff, const void* g) {
    asm volatile("global_load_async_to_lds_b128 %0, %1, off"
                 :: "v"(ldsOff), "v"(g) : "memory");
}

// ---------- fp32 -> bf16 conversion with optional head-scaling ----------
__global__ __launch_bounds__(256) void cvt_kernel(const float* __restrict__ in,
                                                  __bf16* __restrict__ out, int n,
                                                  int nScaled, float f) {
    int i = blockIdx.x * 256 + threadIdx.x;
    if (i < n) {
        float v = in[i];
        if (i < nScaled) v *= f;
        out[i] = (__bf16)v;
    }
}

__global__ __launch_bounds__(256) void bias_scale_kernel(const float* __restrict__ in,
                                                         float* __restrict__ out,
                                                         int n, int nScaled, float f) {
    int i = blockIdx.x * 256 + threadIdx.x;
    if (i < n) out[i] = in[i] * (i < nScaled ? f : 1.0f);
}

// ---------- V transpose: qkv[B,S, 1024 + h*64 + d] -> vt[(bh*64+d)*S + s] ----------
__global__ __launch_bounds__(256) void vtrans_kernel(const __bf16* __restrict__ qkv,
                                                     __bf16* __restrict__ vt) {
    int i = blockIdx.x * 256 + threadIdx.x;   // i < B*S*D_MODEL
    int d = i & 63;
    int h = (i >> 6) & 7;
    int s = (i >> 9) & 4095;
    int b = i >> 21;
    __bf16 v = qkv[(size_t)(b * S_DIM + s) * D3 + 2 * D_MODEL + h * HD + d];
    vt[(size_t)((b * NHEAD + h) * HD + d) * S_DIM + s] = v;
}

// ---------- NT GEMM: C[M,N] = A[M,K] * B[N,K]^T + bias ----------
__global__ __launch_bounds__(256) void gemm_nt_kernel(const __bf16* __restrict__ A,
                                                      const __bf16* __restrict__ Bm,
                                                      const float*  __restrict__ bias,
                                                      __bf16* __restrict__ outB,
                                                      float*  __restrict__ outF,
                                                      int M, int N, int K) {
    const int lane = threadIdx.x & 31;
    const int wave = threadIdx.x >> 5;
    const int id = blockIdx.x * 8 + wave;
    const int ntiles = N >> 4;
    const int tm = id / ntiles;
    const int tn = id - tm * ntiles;
    if (tm * 16 >= M) return;

    const int hi  = lane >> 4;
    const int lo  = lane & 15;
    const int klo = hi * 16;

    const __bf16* arow = A  + (size_t)(tm * 16 + lo) * K;
    const __bf16* brow = Bm + (size_t)(tn * 16 + lo) * K;

    v8f c = {};
    for (int k = 0; k < K; k += 32) {
        __builtin_prefetch(arow + k + 128, 0, 0);
        V16 a;
        a.h[0] = *(const v8bf*)(arow + k + klo);
        a.h[1] = *(const v8bf*)(arow + k + 8 + klo);
        v16bf b = *(const v16bf*)(brow + k + hi * 16);
        c = wmma_bf16(a.v, b, c);
    }

#pragma unroll
    for (int r = 0; r < 8; ++r) {
        int row = tm * 16 + r + 8 * hi;
        int col = tn * 16 + lo;
        float v = c[r] + bias[col];
        if (outB) outB[(size_t)row * N + col] = (__bf16)v;
        if (outF) outF[(size_t)row * N + col] = v;
    }
}

// ---------- Flash attention, inverted-window mask (|i-j|<=win DISALLOWED) ----------
// Block = 4 waves, all same (b,h); wave w owns q-tile qt=(blockIdx&63)*4+w.
// K/V staged cooperatively into LDS with async-to-LDS, double buffered, 64-key chunks.
// LDS tiles padded to 72-element rows (144B) to spread banks.
#define KT0 0            // 2 x 64x72 bf16 = 2 x 9216
#define VT0 18432        // 2 x 64x72 bf16
#define PT0 36864        // 4 x 16x72 bf16 = 4 x 2304
#define SMEM_BYTES 46080

__global__ __launch_bounds__(128) void attn_kernel(const __bf16* __restrict__ qkv,
                                                   const __bf16* __restrict__ vtg,
                                                   const int* __restrict__ winp,
                                                   __bf16* __restrict__ ctx) {
    __shared__ __align__(16) unsigned char smem[SMEM_BYTES];

    const int lane = threadIdx.x & 31;
    const int wave = threadIdx.x >> 5;
    const int bh = blockIdx.x >> 6;                 // 16 (b,h) pairs, 64 blocks each
    const int qt = (blockIdx.x & 63) * 4 + wave;    // 256 q-tiles per bh
    const int b = bh >> 3, h = bh & 7;
    const int q0 = qt * 16;
    const int win = winp[0];

    const int hi  = lane >> 4;
    const int lo  = lane & 15;
    const int klo = hi * 16;

    // Q A-fragments over hd (scale 1/sqrt(hd) already folded into projection)
    const __bf16* qrow = qkv + (size_t)(b * S_DIM + q0 + lo) * D3 + h * HD;
    V16 aq0, aq1;
    aq0.h[0] = *(const v8bf*)(qrow + klo);
    aq0.h[1] = *(const v8bf*)(qrow + 8 + klo);
    aq1.h[0] = *(const v8bf*)(qrow + 32 + klo);
    aq1.h[1] = *(const v8bf*)(qrow + 40 + klo);

    float m[8], l[8];
    v8f o[4];
#pragma unroll
    for (int t = 0; t < 4; ++t) o[t] = {};
#pragma unroll
    for (int r = 0; r < 8; ++r) { m[r] = -1e30f; l[r] = 0.0f; }

    const __bf16* kgbase = qkv + (size_t)(b * S_DIM) * D3 + D_MODEL + h * HD;
    const __bf16* vgbase = vtg + (size_t)(bh * HD) * S_DIM;
    __bf16* pt = (__bf16*)(smem + PT0 + wave * 2304);

    // cooperative stage of one 64-key chunk (K tile + V tile) into buffer `buf`
    auto stage = [&](int k0s, int buf) {
#pragma unroll
        for (int i = 0; i < 4; ++i) {
            int piece = i * 128 + (int)threadIdx.x;   // 512 pieces of 16B per tile
            int row = piece >> 3;                     // key (K tile) / hd (V tile)
            int c8  = (piece & 7) * 8;                // element offset within row
            unsigned ldsK = KT0 + buf * 9216 + (unsigned)(row * 72 + c8) * 2;
            async_ld16(ldsK, kgbase + (size_t)(k0s + row) * D3 + c8);
            unsigned ldsV = VT0 + buf * 9216 + (unsigned)(row * 72 + c8) * 2;
            async_ld16(ldsV, vgbase + (size_t)row * S_DIM + k0s + c8);
        }
    };

    stage(0, 0);
    for (int it = 0; it < 64; ++it) {
        const int bb = it & 1;
        const int k0 = it * 64;
        asm volatile("s_wait_asynccnt 0x0" ::: "memory");
        __syncthreads();                               // buffer bb ready for all waves
        if (it + 1 < 64) stage((it + 1) * 64, bb ^ 1); // overlap DMA with compute

        // wave-uniform chunk classification vs banned band |i-j|<=win
        const bool skipAll = ((k0 + 63 - q0) <= win) && ((q0 + 15 - k0) <= win);
        if (!skipAll) {
            const bool partial = !((k0 - (q0 + 15) > win) || (q0 - (k0 + 63) > win));
            const __bf16* ktb = (const __bf16*)(smem + KT0 + bb * 9216);
            const __bf16* vtb = (const __bf16*)(smem + VT0 + bb * 9216);

            // ---- scores: 4 key tiles x (hd contraction 64 -> 2 WMMAs)
            v8f s[4];
#pragma unroll
            for (int t = 0; t < 4; ++t) {
                v8f st = {};
                const __bf16* kr = ktb + (16 * t + lo) * 72 + hi * 16;
                st = wmma_bf16(aq0.v, *(const v16bf*)(kr), st);
                st = wmma_bf16(aq1.v, *(const v16bf*)(kr + 32), st);
                s[t] = st;
            }

            // ---- online softmax over 64 keys
#pragma unroll
            for (int r = 0; r < 8; ++r) {
                int qr = q0 + r + 8 * hi;
                float v0 = s[0][r], v1 = s[1][r], v2 = s[2][r], v3 = s[3][r];
                if (partial) {
                    int d0 = qr - (k0 + lo);      if (d0 < 0) d0 = -d0;
                    int d1 = qr - (k0 + 16 + lo); if (d1 < 0) d1 = -d1;
                    int d2 = qr - (k0 + 32 + lo); if (d2 < 0) d2 = -d2;
                    int d3 = qr - (k0 + 48 + lo); if (d3 < 0) d3 = -d3;
                    if (d0 <= win) v0 = -1e30f;
                    if (d1 <= win) v1 = -1e30f;
                    if (d2 <= win) v2 = -1e30f;
                    if (d3 <= win) v3 = -1e30f;
                }
                float mrow = rmax16(fmaxf(fmaxf(v0, v1), fmaxf(v2, v3)));
                float mnew = fmaxf(m[r], mrow);
                float alpha = __expf(m[r] - mnew);
                float p0 = __expf(v0 - mnew);
                float p1 = __expf(v1 - mnew);
                float p2 = __expf(v2 - mnew);
                float p3 = __expf(v3 - mnew);
                l[r] = l[r] * alpha + rsum16((p0 + p1) + (p2 + p3));
                m[r] = mnew;
                o[0][r] *= alpha; o[1][r] *= alpha; o[2][r] *= alpha; o[3][r] *= alpha;
                int row = r + 8 * hi;
                pt[row * 72 + lo]      = (__bf16)p0;
                pt[row * 72 + 16 + lo] = (__bf16)p1;
                pt[row * 72 + 32 + lo] = (__bf16)p2;
                pt[row * 72 + 48 + lo] = (__bf16)p3;
            }

            asm volatile("s_wait_dscnt 0x0" ::: "memory");  // P visible to A-frag reads

            V16 ap0, ap1;                                   // P 16x64 -> two A frags
            ap0.h[0] = *(const v8bf*)(pt + lo * 72 + klo);
            ap0.h[1] = *(const v8bf*)(pt + lo * 72 + 8 + klo);
            ap1.h[0] = *(const v8bf*)(pt + lo * 72 + 32 + klo);
            ap1.h[1] = *(const v8bf*)(pt + lo * 72 + 40 + klo);

            // ---- O += P(16x64) x V(64x64)
#pragma unroll
            for (int t = 0; t < 4; ++t) {
                const __bf16* vr = vtb + (16 * t + lo) * 72 + hi * 16;
                o[t] = wmma_bf16(ap0.v, *(const v16bf*)(vr), o[t]);
                o[t] = wmma_bf16(ap1.v, *(const v16bf*)(vr + 32), o[t]);
            }
        }
        __syncthreads();   // all waves done reading bb before it is re-staged
    }

    // ---- normalize and write ctx (bf16, [B*S, 512], col = h*64 + d)
#pragma unroll
    for (int r = 0; r < 8; ++r) {
        float inv = 1.0f / l[r];
        int q = q0 + r + 8 * hi;
        __bf16* crow = ctx + (size_t)(b * S_DIM + q) * D_MODEL + h * HD + lo;
        crow[0]  = (__bf16)(o[0][r] * inv);
        crow[16] = (__bf16)(o[1][r] * inv);
        crow[32] = (__bf16)(o[2][r] * inv);
        crow[48] = (__bf16)(o[3][r] * inv);
    }
}

extern "C" void kernel_launch(void* const* d_in, const int* in_sizes, int n_in,
                              void* d_out, int out_size, void* d_ws, size_t ws_size,
                              hipStream_t stream) {
    const float* x   = (const float*)d_in[0];   // [2,4096,512]
    const float* ipw = (const float*)d_in[1];   // [1536,512]
    const float* ipb = (const float*)d_in[2];   // [1536]
    const float* opw = (const float*)d_in[3];   // [512,512]
    const float* opb = (const float*)d_in[4];   // [512]
    const int*   win = (const int*)d_in[5];     // scalar

    char* p = (char*)d_ws;
    __bf16* Xb  = (__bf16*)(p);                         //  8388608 B (8192x512)
    __bf16* Wq  = (__bf16*)(p + 8388608);               //  1572864 B (1536x512)
    __bf16* Wo  = (__bf16*)(p + 9961472);               //   524288 B (512x512)
    __bf16* QKV = (__bf16*)(p + 10485760);              // 25165824 B (8192x1536)
    __bf16* Vt  = (__bf16*)(p + 35651584);              //  8388608 B (16x64x4096)
    __bf16* Ctx = (__bf16*)(p + 44040192);              //  8388608 B (8192x512)
    float*  Bs  = (float*)(p + 52428800);               //     6144 B (scaled in_proj_b)

    const float qscale = 0.125f;   // 1/sqrt(64)

    // 1) conversions; fold q-scale into Q rows of in_proj (rows 0..511) and bias
    cvt_kernel<<<16384, 256, 0, stream>>>(x,   Xb, B_DIM * S_DIM * D_MODEL, 0, 1.0f);
    cvt_kernel<<<3072,  256, 0, stream>>>(ipw, Wq, D3 * D_MODEL,
                                          D_MODEL * D_MODEL, qscale);
    cvt_kernel<<<1024,  256, 0, stream>>>(opw, Wo, D_MODEL * D_MODEL, 0, 1.0f);
    bias_scale_kernel<<<6, 256, 0, stream>>>(ipb, Bs, D3, D_MODEL, qscale);

    // 2) QKV projection: [8192,512] x [1536,512]^T + b -> bf16 [8192,1536]
    gemm_nt_kernel<<<6144, 256, 0, stream>>>(Xb, Wq, Bs, QKV, (float*)nullptr,
                                             B_DIM * S_DIM, D3, D_MODEL);

    // 3) transpose V into [bh,hd,S]
    vtrans_kernel<<<16384, 256, 0, stream>>>(QKV, Vt);

    // 4) flash attention: 16 bh x 64 blocks x 4 waves
    attn_kernel<<<1024, 128, 0, stream>>>(QKV, Vt, win, Ctx);

    // 5) output projection: [8192,512] x [512,512]^T + b -> fp32 d_out
    gemm_nt_kernel<<<2048, 256, 0, stream>>>(Ctx, Wo, opb, (__bf16*)nullptr,
                                             (float*)d_out, B_DIM * S_DIM,
                                             D_MODEL, D_MODEL);
}
